// SparseResBlock3d_21792664060178
// MI455X (gfx1250) — matrix operand order repaired and checked by reading, not compiled, not run on hardware
//
#include <hip/hip_runtime.h>

typedef __attribute__((ext_vector_type(16))) _Float16 v16h;
typedef __attribute__((ext_vector_type(8)))  _Float16 v8h;
typedef __attribute__((ext_vector_type(8)))  float    v8f;

#define N_VOX 100000
#define C_IN  32
#define C_OUT 64
#define KOFF  27
#define N_TILES (N_VOX / 16)       // 6250, exact
#define N_PAIRS (N_TILES / 2)      // 3125: 2 M-tiles (32 voxels) per wave
#define LN_EPS 1e-6f

#define WMMA_F16(A, B, C) \
    __builtin_amdgcn_wmma_f32_16x16x32_f16(false, (A), false, (B), (short)0, (C), false, false)

// ---------------------------------------------------------------------------
// Weight repack: f32 [27,Cin,Cout] -> f16 WMMA B-fragments.
// B-frag layout (16x16x32 f16, 32x16 KxN): lane holds column n = lane&15,
// K index = (lane>>4)*16 + p for half-position p in 0..15 (this code's order, same as A; ISA table order differs).
// ---------------------------------------------------------------------------
__global__ void pack_w1_kernel(const float* __restrict__ W1, _Float16* __restrict__ pw1) {
    int tid = blockIdx.x * 256 + threadIdx.x;          // 27*4*32*16 = 55296
    if (tid >= KOFF * 4 * 32 * 16) return;
    int p     = tid & 15;
    int lane  = (tid >> 4) & 31;
    int ntile = (tid >> 9) & 3;
    int k     = tid >> 11;
    int kidx  = ((lane >> 4) << 4) + p;                // 0..31
    int n     = ntile * 16 + (lane & 15);              // 0..63
    pw1[tid] = (_Float16)W1[(k * C_IN + kidx) * C_OUT + n];
}

__global__ void pack_w2_kernel(const float* __restrict__ W2, _Float16* __restrict__ pw2) {
    int tid = blockIdx.x * 256 + threadIdx.x;          // 27*2*4*32*16 = 110592
    if (tid >= KOFF * 2 * 4 * 32 * 16) return;
    int p     = tid & 15;
    int lane  = (tid >> 4) & 31;
    int ntile = (tid >> 9) & 3;
    int s     = (tid >> 11) & 1;                       // K-step within 64
    int k     = tid >> 12;
    int kidx  = s * 32 + ((lane >> 4) << 4) + p;       // 0..63
    int n     = ntile * 16 + (lane & 15);
    pw2[tid] = (_Float16)W2[(k * C_OUT + kidx) * C_OUT + n];
}

// ---------------------------------------------------------------------------
// Remap neighbor table: -1 -> N_VOX (points at an all-zero feature row),
// making the conv inner loops branch-free (EXEC stays all-1s trivially).
// ---------------------------------------------------------------------------
__global__ void remap_nbr_kernel(const int* __restrict__ nbr, int* __restrict__ nbr2) {
    int i = blockIdx.x * 256 + threadIdx.x;
    if (i >= N_VOX * KOFF) return;
    int v = nbr[i];
    nbr2[i] = (v < 0) ? N_VOX : v;
}

__global__ void zero_rows_kernel(_Float16* __restrict__ h1, _Float16* __restrict__ h2) {
    int t = threadIdx.x;
    if (t < C_IN)  h1[N_VOX * C_IN  + t] = (_Float16)0.f;
    if (t < C_OUT) h2[N_VOX * C_OUT + t] = (_Float16)0.f;
}

// ---------------------------------------------------------------------------
// LN(gamma,beta) + SiLU on input feats -> h1 (f16 [N+1,32])
// ---------------------------------------------------------------------------
__global__ void ln_silu_in_kernel(const float* __restrict__ feats,
                                  const float* __restrict__ gamma,
                                  const float* __restrict__ beta,
                                  _Float16* __restrict__ h1) {
    int v = blockIdx.x * 256 + threadIdx.x;
    if (v >= N_VOX) return;
    const float* row = feats + v * C_IN;
    float sum = 0.f, sumsq = 0.f;
    #pragma unroll
    for (int c = 0; c < C_IN; ++c) { float x = row[c]; sum += x; sumsq += x * x; }
    float mu  = sum * (1.f / C_IN);
    float var = sumsq * (1.f / C_IN) - mu * mu;
    float r   = rsqrtf(var + LN_EPS);
    #pragma unroll
    for (int c = 0; c < C_IN; ++c) {
        float y = (row[c] - mu) * r * gamma[c] + beta[c];
        float s = y / (1.f + __expf(-y));
        h1[v * C_IN + c] = (_Float16)s;
    }
}

// ---------------------------------------------------------------------------
// LN (no affine) + SiLU on conv1 output t1 -> h2 (f16 [N+1,64])
// ---------------------------------------------------------------------------
__global__ void ln_silu_mid_kernel(const float* __restrict__ t1,
                                   _Float16* __restrict__ h2) {
    int v = blockIdx.x * 256 + threadIdx.x;
    if (v >= N_VOX) return;
    const float* row = t1 + v * C_OUT;
    float sum = 0.f, sumsq = 0.f;
    #pragma unroll
    for (int c = 0; c < C_OUT; ++c) { float x = row[c]; sum += x; sumsq += x * x; }
    float mu  = sum * (1.f / C_OUT);
    float var = sumsq * (1.f / C_OUT) - mu * mu;
    float r   = rsqrtf(var + LN_EPS);
    #pragma unroll
    for (int c = 0; c < C_OUT; ++c) {
        float y = (row[c] - mu) * r;
        float s = y / (1.f + __expf(-y));
        h2[v * C_OUT + c] = (_Float16)s;
    }
}

// ---------------------------------------------------------------------------
// Skip path: out[n,d] = feats[n,:] . Wskip[:,d] + bskip[d]
// ---------------------------------------------------------------------------
__global__ void skip_gemm_kernel(const float* __restrict__ feats,
                                 const float* __restrict__ Wskip,
                                 const float* __restrict__ bskip,
                                 float* __restrict__ out) {
    int tid = blockIdx.x * 256 + threadIdx.x;
    if (tid >= N_VOX * C_OUT) return;
    int v = tid >> 6, d = tid & 63;
    const float* f = feats + v * C_IN;
    float s = bskip[d];
    #pragma unroll
    for (int c = 0; c < C_IN; ++c) s += f[c] * Wskip[c * C_OUT + d];
    out[tid] = s;
}

// ---------------------------------------------------------------------------
// conv1: one wave per TWO 16-voxel tiles (32 voxels x 64 out ch).
// Each B fragment feeds 2 WMMAs. Branch-free A loads via nbr2 + zero row.
// A-frag layout (16x32 f16 MxK): row m = lane&15,
//   K = (lane>>4)*16 + p (this code's order, as in pw1) => two contiguous 16B chunks at
//   byte offsets (lane>>4)*32 and 16+(lane>>4)*32 of the neighbor row.
// ---------------------------------------------------------------------------
__global__ void conv1_wmma_kernel(const _Float16* __restrict__ h1,
                                  const int* __restrict__ nbr2,
                                  const _Float16* __restrict__ pw1,
                                  const float* __restrict__ bconv1,
                                  float* __restrict__ t1) {
    int gtid = blockIdx.x * blockDim.x + threadIdx.x;
    int wave = gtid >> 5;
    int lane = threadIdx.x & 31;
    if (wave >= N_PAIRS) return;                       // wave-uniform
    int base = wave * 32;
    int half = lane >> 4;
    int m0   = base + (lane & 15);
    const int* nb0 = nbr2 + m0 * KOFF;
    const int* nb1 = nb0 + 16 * KOFF;                  // m1 = m0 + 16

    v8f acc[8] = {};
    const v16h* pwv = (const v16h*)pw1;

    int idx0 = nb0[0], idx1 = nb1[0];
    for (int k = 0; k < KOFF; ++k) {
        const _Float16* hp0 = h1 + idx0 * C_IN + half * 16;
        const _Float16* hp1 = h1 + idx1 * C_IN + half * 16;
        union { v16h v; v8h h[2]; } a0, a1;
        a0.h[0] = *(const v8h*)hp0;  a0.h[1] = *(const v8h*)(hp0 + 8);
        a1.h[0] = *(const v8h*)hp1;  a1.h[1] = *(const v8h*)(hp1 + 8);
        const v16h* bk = pwv + (k * 128 + lane);
        v16h b0 = bk[0], b1 = bk[32], b2 = bk[64], b3 = bk[96];
        if (k + 1 < KOFF) { idx0 = nb0[k + 1]; idx1 = nb1[k + 1]; }  // prefetch
        acc[0] = WMMA_F16(a0.v, b0, acc[0]);
        acc[4] = WMMA_F16(a1.v, b0, acc[4]);
        acc[1] = WMMA_F16(a0.v, b1, acc[1]);
        acc[5] = WMMA_F16(a1.v, b1, acc[5]);
        acc[2] = WMMA_F16(a0.v, b2, acc[2]);
        acc[6] = WMMA_F16(a1.v, b2, acc[6]);
        acc[3] = WMMA_F16(a0.v, b3, acc[3]);
        acc[7] = WMMA_F16(a1.v, b3, acc[7]);
    }

    // C/D layout: lane holds N = lane&15, M = (lane>>4)*8 + vreg
    int col = lane & 15;
    #pragma unroll
    for (int g = 0; g < 2; ++g) {                      // two M-tiles
        int rowbase = base + g * 16 + half * 8;
        #pragma unroll
        for (int t = 0; t < 4; ++t) {
            float b = bconv1[t * 16 + col];
            #pragma unroll
            for (int r = 0; r < 8; ++r)
                t1[(rowbase + r) * C_OUT + t * 16 + col] = acc[g * 4 + t][r] + b;
        }
    }
}

// ---------------------------------------------------------------------------
// conv2: K = 27x64 => two 16x16x32 K-steps per offset; accumulate into out.
// ---------------------------------------------------------------------------
__global__ void conv2_wmma_kernel(const _Float16* __restrict__ h2,
                                  const int* __restrict__ nbr2,
                                  const _Float16* __restrict__ pw2,
                                  const float* __restrict__ bconv2,
                                  float* __restrict__ out) {
    int gtid = blockIdx.x * blockDim.x + threadIdx.x;
    int wave = gtid >> 5;
    int lane = threadIdx.x & 31;
    if (wave >= N_PAIRS) return;
    int base = wave * 32;
    int half = lane >> 4;
    int m0   = base + (lane & 15);
    const int* nb0 = nbr2 + m0 * KOFF;
    const int* nb1 = nb0 + 16 * KOFF;

    v8f acc[8] = {};
    const v16h* pwv = (const v16h*)pw2;

    int idx0 = nb0[0], idx1 = nb1[0];
    for (int k = 0; k < KOFF; ++k) {
        const _Float16* r0 = h2 + idx0 * C_OUT + half * 16;
        const _Float16* r1 = h2 + idx1 * C_OUT + half * 16;
        if (k + 1 < KOFF) { idx0 = nb0[k + 1]; idx1 = nb1[k + 1]; }  // prefetch
        #pragma unroll
        for (int s = 0; s < 2; ++s) {
            union { v16h v; v8h h[2]; } a0, a1;
            const _Float16* hp0 = r0 + s * 32;
            const _Float16* hp1 = r1 + s * 32;
            a0.h[0] = *(const v8h*)hp0;  a0.h[1] = *(const v8h*)(hp0 + 8);
            a1.h[0] = *(const v8h*)hp1;  a1.h[1] = *(const v8h*)(hp1 + 8);
            const v16h* bk = pwv + ((k * 2 + s) * 128 + lane);
            v16h b0 = bk[0], b1 = bk[32], b2 = bk[64], b3 = bk[96];
            acc[0] = WMMA_F16(a0.v, b0, acc[0]);
            acc[4] = WMMA_F16(a1.v, b0, acc[4]);
            acc[1] = WMMA_F16(a0.v, b1, acc[1]);
            acc[5] = WMMA_F16(a1.v, b1, acc[5]);
            acc[2] = WMMA_F16(a0.v, b2, acc[2]);
            acc[6] = WMMA_F16(a1.v, b2, acc[6]);
            acc[3] = WMMA_F16(a0.v, b3, acc[3]);
            acc[7] = WMMA_F16(a1.v, b3, acc[7]);
        }
    }

    int col = lane & 15;
    #pragma unroll
    for (int g = 0; g < 2; ++g) {
        int rowbase = base + g * 16 + half * 8;
        #pragma unroll
        for (int t = 0; t < 4; ++t) {
            float b = bconv2[t * 16 + col];
            #pragma unroll
            for (int r = 0; r < 8; ++r) {
                int o = (rowbase + r) * C_OUT + t * 16 + col;
                out[o] = out[o] + acc[g * 4 + t][r] + b;   // skip already in out
            }
        }
    }
}

// ---------------------------------------------------------------------------
extern "C" void kernel_launch(void* const* d_in, const int* in_sizes, int n_in,
                              void* d_out, int out_size, void* d_ws, size_t ws_size,
                              hipStream_t stream) {
    const float* feats  = (const float*)d_in[0];
    const int*   nbr    = (const int*)d_in[1];   // int32 after JAX downcast
    const float* gamma1 = (const float*)d_in[2];
    const float* beta1  = (const float*)d_in[3];
    const float* W1     = (const float*)d_in[4];
    const float* bconv1 = (const float*)d_in[5];
    const float* W2     = (const float*)d_in[6];
    const float* bconv2 = (const float*)d_in[7];
    const float* Wskip  = (const float*)d_in[8];
    const float* bskip  = (const float*)d_in[9];
    float* out = (float*)d_out;

    char* ws = (char*)d_ws;
    size_t o = 0;
    _Float16* pw1  = (_Float16*)(ws + o); o += 131072;       // 110592 B used
    _Float16* pw2  = (_Float16*)(ws + o); o += 262144;       // 221184 B used
    _Float16* h1   = (_Float16*)(ws + o); o += 6422528;      // (N+1)*32*2
    _Float16* h2   = (_Float16*)(ws + o); o += 12845056;     // (N+1)*64*2
    int*      nbr2 = (int*)     (ws + o); o += 10800128;     // N*27*4
    float*    t1   = (float*)   (ws + o);                    // N*64*4

    pack_w1_kernel<<<(KOFF * 4 * 32 * 16 + 255) / 256, 256, 0, stream>>>(W1, pw1);
    pack_w2_kernel<<<(KOFF * 2 * 4 * 32 * 16 + 255) / 256, 256, 0, stream>>>(W2, pw2);
    remap_nbr_kernel<<<(N_VOX * KOFF + 255) / 256, 256, 0, stream>>>(nbr, nbr2);
    zero_rows_kernel<<<1, 64, 0, stream>>>(h1, h2);
    ln_silu_in_kernel<<<(N_VOX + 255) / 256, 256, 0, stream>>>(feats, gamma1, beta1, h1);
    skip_gemm_kernel<<<(N_VOX * C_OUT + 255) / 256, 256, 0, stream>>>(feats, Wskip, bskip, out);
    conv1_wmma_kernel<<<(N_PAIRS * 32 + 255) / 256, 256, 0, stream>>>(h1, nbr2, pw1, bconv1, t1);
    ln_silu_mid_kernel<<<(N_VOX + 255) / 256, 256, 0, stream>>>(t1, h2);
    conv2_wmma_kernel<<<(N_PAIRS * 32 + 255) / 256, 256, 0, stream>>>(h2, nbr2, pw2, bconv2, out);
}
